// CrossAttention_11862699672049
// MI455X (gfx1250) — matrix operand order repaired and checked by reading, not compiled, not run on hardware
//
#include <hip/hip_runtime.h>

typedef _Float16 h16v __attribute__((ext_vector_type(16)));
typedef _Float16 h8v  __attribute__((ext_vector_type(8)));
typedef _Float16 h4v  __attribute__((ext_vector_type(4)));
typedef float    f8v  __attribute__((ext_vector_type(8)));
typedef float    f4v  __attribute__((ext_vector_type(4)));

// ---------------------------------------------------------------------------
// CDNA5 async global->LDS copy (16 bytes per lane), tracked by ASYNCcnt.
// Flat pointers to __shared__ carry the LDS byte offset in addr[31:0].
// ---------------------------------------------------------------------------
__device__ __forceinline__ void async_copy_b128(void* lds_dst, const void* gsrc) {
  unsigned int loff = (unsigned int)(uintptr_t)lds_dst;
  asm volatile("global_load_async_to_lds_b128 %0, %1, off"
               :: "v"(loff), "v"(gsrc) : "memory");
}
__device__ __forceinline__ void wait_asynccnt0() {
  asm volatile("s_wait_asynccnt 0" ::: "memory");
}

// ---------------------------------------------------------------------------
// Fragment loaders (CDNA5 wave32 WMMA layouts, 16x16x32 f16)
// ---------------------------------------------------------------------------

// A matrix 16x32 (MxK), row-major source with row stride `ld` (halfs).
// lane<16: row=lane, K={0..7,16..23}; lane>=16: row=lane-16, K={8..15,24..31}
__device__ __forceinline__ h16v load_afrag(const _Float16* p, int ld, int lane) {
  const int m  = lane & 15;
  const int hi = (lane >> 4) & 1;
  const _Float16* r = p + (size_t)m * ld + hi * 8;
  h8v lo = *(const h8v*)(r);
  h8v hh = *(const h8v*)(r + 16);
  h16v v;
#pragma unroll
  for (int i = 0; i < 8; ++i) { v[i] = lo[i]; v[i + 8] = hh[i]; }
  return v;
}

// B matrix 32x16 (KxN). Source laid out "n-major, k-contiguous":
// row n holds the 32 K-values for column n. lane<16: K={0..7,16..23}; lane>=16: K={8..15,24..31}
__device__ __forceinline__ h16v load_bfrag(const _Float16* p, int ld, int lane) {
  const int n    = lane & 15;
  const int koff = (lane & 16) ? 8 : 0;
  const _Float16* r = p + (size_t)n * ld + koff;
  h8v lo = *(const h8v*)(r);
  h8v hh = *(const h8v*)(r + 16);
  h16v v;
#pragma unroll
  for (int i = 0; i < 8; ++i) { v[i] = lo[i]; v[i + 8] = hh[i]; }
  return v;
}

// ---------------------------------------------------------------------------
// Generic GEMM: C[M,N] = A[M,K] @ B[K,N]
//   A: f32 (A_HALF=0, VALU convert) or f16 (A_HALF=1, async-DMA to LDS)
//   B: f32 row-major (converted to f16, stored transposed)
//   Out: f16 scaled (OUT_BIAS_F32=0) or f32 + bias (OUT_BIAS_F32=1)
// Block: 256 threads = 8 waves, tile 64(M) x 128(N), K staged 64 per step.
// ---------------------------------------------------------------------------
template <bool A_HALF, bool OUT_BIAS_F32>
__global__ __launch_bounds__(256) void gemm_wmma(
    const void* __restrict__ Ap, const float* __restrict__ Bg,
    const float* __restrict__ bias, void* __restrict__ Cp,
    int M, int N, int K, float outScale) {
  __shared__ __attribute__((aligned(16))) _Float16 As[64][72];   // [m][k]
  __shared__ __attribute__((aligned(16))) _Float16 Bs[128][72];  // [n][k] (B^T)

  const int tid  = threadIdx.x;
  const int lane = tid & 31;
  const int wid  = tid >> 5;
  const int wM   = wid & 3;   // 4 waves along M
  const int wN   = wid >> 2;  // 2 waves along N
  const int m0   = blockIdx.x * 64;
  const int n0   = blockIdx.y * 128;

  const float*    Af = (const float*)Ap;
  const _Float16* Ah = (const _Float16*)Ap;

  const f8v z8 = {0.f,0.f,0.f,0.f,0.f,0.f,0.f,0.f};
  f8v acc[4];
#pragma unroll
  for (int t = 0; t < 4; ++t) acc[t] = z8;

  for (int k0 = 0; k0 < K; k0 += 64) {
    if (A_HALF) {
      // async DMA: 64x64 halfs = 512 x 16B chunks, 2 per thread
#pragma unroll
      for (int e = 0; e < 2; ++e) {
        int c = e * 256 + tid;
        int mm = c >> 3, seg = c & 7;
        async_copy_b128(&As[mm][seg * 8],
                        Ah + (size_t)(m0 + mm) * K + k0 + seg * 8);
      }
    } else {
      // f32 -> f16: float4 loads, packed b64 LDS stores; 4 chunks per thread
#pragma unroll
      for (int e = 0; e < 4; ++e) {
        int c = e * 256 + tid;
        int mm = c >> 4, k4 = c & 15;
        f4v v = *(const f4v*)(Af + (size_t)(m0 + mm) * K + k0 + k4 * 4);
        h4v h; 
#pragma unroll
        for (int i = 0; i < 4; ++i) h[i] = (_Float16)v[i];
        *(h4v*)&As[mm][k4 * 4] = h;
      }
    }
    // stage B block 64x128 transposed -> Bs[n][k]; float4 loads, b16 scatter
#pragma unroll
    for (int e = 0; e < 8; ++e) {
      int c = e * 256 + tid;
      int kk = c >> 5, n4 = c & 31;
      f4v v = *(const f4v*)(Bg + (size_t)(k0 + kk) * N + n0 + n4 * 4);
#pragma unroll
      for (int i = 0; i < 4; ++i) Bs[n4 * 4 + i][kk] = (_Float16)v[i];
    }
    if (A_HALF) wait_asynccnt0();
    __syncthreads();

#pragma unroll
    for (int ks = 0; ks < 2; ++ks) {
      h16v a = load_afrag(&As[16 * wM][32 * ks], 72, lane);
#pragma unroll
      for (int t = 0; t < 4; ++t) {
        h16v b = load_bfrag(&Bs[64 * wN + 16 * t][32 * ks], 72, lane);
        acc[t] = __builtin_amdgcn_wmma_f32_16x16x32_f16(
            false, a, false, b, (short)0, acc[t], false, false);
      }
    }
    __syncthreads();
  }

  const int hi = (lane >> 4) & 1;
  const int nl = lane & 15;
#pragma unroll
  for (int t = 0; t < 4; ++t) {
#pragma unroll
    for (int r = 0; r < 8; ++r) {
      int row = m0 + 16 * wM + r + 8 * hi;
      int col = n0 + 64 * wN + 16 * t + nl;
      if (OUT_BIAS_F32) {
        ((float*)Cp)[(size_t)row * N + col] = acc[t][r] + bias[col];
      } else {
        ((_Float16*)Cp)[(size_t)row * N + col] = (_Float16)(acc[t][r] * outScale);
      }
    }
  }
}

// ---------------------------------------------------------------------------
// Flash attention: one block = 128 threads (4 waves) per (b, h, 64-row q tile).
// Each wave owns 16 query rows; KV iterated in 64-row chunks, double-buffered:
// K chunk i+1 streams into LDS via the async engine while chunk i is consumed
// by WMMAs; V^T chunk i+1 staged (vectorized, transposing) in the same slot.
// ---------------------------------------------------------------------------
__global__ __launch_bounds__(128) void attn_wmma(
    const _Float16* __restrict__ Qh,   // [8192, 1024]
    const _Float16* __restrict__ KVh,  // [2048, 2048] k: h*64+d, v: 1024+h*64+d
    _Float16* __restrict__ Oh) {       // [8192, 1024]
  __shared__ __attribute__((aligned(16))) _Float16 Ks[2][64][72];    // [kv][d]
  __shared__ __attribute__((aligned(16))) _Float16 Vs[2][64][72];    // [d][kv]
  __shared__ __attribute__((aligned(16))) _Float16 Ps[4][16][72];    // per-wave P

  const int tid  = threadIdx.x;
  const int lane = tid & 31;
  const int wid  = tid >> 5;

  const int qt = blockIdx.x & 15;
  const int h  = (blockIdx.x >> 4) & 15;
  const int b  = blockIdx.x >> 8;

  const int qrow0 = b * 1024 + qt * 64 + wid * 16;
  const int hcol  = h * 64;
  const int hi = (lane >> 4) & 1;
  const int nl = lane & 15;

  // Q fragments for the wave's 16 rows (D=64 -> 2 k-steps); scale pre-folded
  h16v aq[2];
#pragma unroll
  for (int ks = 0; ks < 2; ++ks)
    aq[ks] = load_afrag(Qh + (size_t)qrow0 * 1024 + hcol + 32 * ks, 1024, lane);

  const f8v z8 = {0.f,0.f,0.f,0.f,0.f,0.f,0.f,0.f};
  f8v o[4];
#pragma unroll
  for (int t = 0; t < 4; ++t) o[t] = z8;
  float mi[8], li[8];
#pragma unroll
  for (int r = 0; r < 8; ++r) { mi[r] = -1e30f; li[r] = 0.f; }

  // K staging: async DMA, 64 rows x 128B = 512 x 16B chunks, 4 per thread
  auto stageK = [&](int buf, int j0) {
#pragma unroll
    for (int e = 0; e < 4; ++e) {
      int c = e * 128 + tid;
      int jj = c >> 3, seg = c & 7;
      async_copy_b128(&Ks[buf][jj][seg * 8],
                      KVh + (size_t)(j0 + jj) * 2048 + hcol + seg * 8);
    }
  };
  // V^T staging: b128 global reads + transposing b16 LDS scatter
  auto stageV = [&](int buf, int j0) {
#pragma unroll
    for (int e = 0; e < 4; ++e) {
      int c = e * 128 + tid;
      int jj = c >> 3, seg = c & 7;
      h8v v = *(const h8v*)(KVh + (size_t)(j0 + jj) * 2048 + 1024 + hcol + seg * 8);
#pragma unroll
      for (int i = 0; i < 8; ++i) Vs[buf][seg * 8 + i][jj] = v[i];
    }
  };

  stageK(0, 0);
  stageV(0, 0);
  wait_asynccnt0();
  __syncthreads();

  for (int it = 0; it < 32; ++it) {
    const int cur = it & 1;
    // prefetch next chunk into the other buffer (overlaps with WMMAs below)
    if (it + 1 < 32) {
      stageK(cur ^ 1, (it + 1) * 64);
      stageV(cur ^ 1, (it + 1) * 64);
    }

    // S = Q K^T  (16 x 64 per wave)
    f8v s[4];
#pragma unroll
    for (int t = 0; t < 4; ++t) s[t] = z8;
#pragma unroll
    for (int ks = 0; ks < 2; ++ks) {
#pragma unroll
      for (int t = 0; t < 4; ++t) {
        h16v bk = load_bfrag(&Ks[cur][16 * t][32 * ks], 72, lane);
        s[t] = __builtin_amdgcn_wmma_f32_16x16x32_f16(
            false, aq[ks], false, bk, (short)0, s[t], false, false);
      }
    }

    // online softmax: rows live across 16-lane groups at fixed accum index r
#pragma unroll
    for (int r = 0; r < 8; ++r) {
      float mx = fmaxf(fmaxf(s[0][r], s[1][r]), fmaxf(s[2][r], s[3][r]));
#pragma unroll
      for (int off = 1; off < 16; off <<= 1)
        mx = fmaxf(mx, __shfl_xor(mx, off, 32));
      float mnew  = fmaxf(mi[r], mx);
      float alpha = __expf(mi[r] - mnew);
      mi[r] = mnew;
      float rs = 0.f;
#pragma unroll
      for (int t = 0; t < 4; ++t) {
        float p = __expf(s[t][r] - mnew);
        s[t][r] = p;
        rs += p;
      }
#pragma unroll
      for (int off = 1; off < 16; off <<= 1)
        rs += __shfl_xor(rs, off, 32);
      li[r] = li[r] * alpha + rs;
#pragma unroll
      for (int t = 0; t < 4; ++t) o[t][r] *= alpha;
    }

    // P: D-layout -> per-wave LDS tile -> A-layout
#pragma unroll
    for (int t = 0; t < 4; ++t)
#pragma unroll
      for (int r = 0; r < 8; ++r)
        Ps[wid][r + 8 * hi][16 * t + nl] = (_Float16)s[t][r];

    h16v ap[2];
#pragma unroll
    for (int ks = 0; ks < 2; ++ks)
      ap[ks] = load_afrag(&Ps[wid][0][32 * ks], 72, lane);

    // O += P @ V
#pragma unroll
    for (int t = 0; t < 4; ++t) {
#pragma unroll
      for (int ks = 0; ks < 2; ++ks) {
        h16v bv = load_bfrag(&Vs[cur][16 * t][32 * ks], 72, lane);
        o[t] = __builtin_amdgcn_wmma_f32_16x16x32_f16(
            false, ap[ks], false, bv, (short)0, o[t], false, false);
      }
    }

    wait_asynccnt0();   // this wave's next-chunk K DMA done
    __syncthreads();    // all waves' staging visible; safe to swap buffers
  }

  // normalize and store f16 attention output
#pragma unroll
  for (int r = 0; r < 8; ++r) {
    float rl = 1.f / li[r];
    int row = qrow0 + r + 8 * hi;
#pragma unroll
    for (int t = 0; t < 4; ++t)
      Oh[(size_t)row * 1024 + hcol + 16 * t + nl] = (_Float16)(o[t][r] * rl);
  }
}

// ---------------------------------------------------------------------------
extern "C" void kernel_launch(void* const* d_in, const int* in_sizes, int n_in,
                              void* d_out, int out_size, void* d_ws, size_t ws_size,
                              hipStream_t stream) {
  const float* x       = (const float*)d_in[0];  // [8,1024,1024]
  const float* context = (const float*)d_in[1];  // [2048,1024]
  const float* q_w     = (const float*)d_in[2];  // [1024,1024]
  const float* kv_w    = (const float*)d_in[3];  // [1024,2048]
  const float* proj_w  = (const float*)d_in[4];  // [1024,1024]
  const float* proj_b  = (const float*)d_in[5];  // [1024]

  _Float16* qh  = (_Float16*)d_ws;               // [8192,1024] f16, 16 MB
  _Float16* kvh = qh  + (size_t)8192 * 1024;     // [2048,2048] f16,  8 MB
  _Float16* oh  = kvh + (size_t)2048 * 2048;     // [8192,1024] f16, 16 MB

  // Q = (x @ q_w) * D^-0.5   (scale folded into f16 output)
  gemm_wmma<false, false><<<dim3(8192 / 64, 1024 / 128), 256, 0, stream>>>(
      x, q_w, nullptr, (void*)qh, 8192, 1024, 1024, 0.125f);
  // KV = context @ kv_w
  gemm_wmma<false, false><<<dim3(2048 / 64, 2048 / 128), 256, 0, stream>>>(
      context, kv_w, nullptr, (void*)kvh, 2048, 2048, 1024, 1.0f);
  // flash attention: B*H*(Nq/64) = 2048 blocks
  attn_wmma<<<dim3(2048), 128, 0, stream>>>(qh, kvh, oh);
  // out = O @ proj_w + proj_b  (f32 output)
  gemm_wmma<true, true><<<dim3(8192 / 64, 1024 / 128), 256, 0, stream>>>(
      (const void*)oh, proj_w, proj_b, d_out, 8192, 1024, 1024, 1.0f);
  (void)in_sizes; (void)n_in; (void)out_size; (void)ws_size;
}
